// Mix9Net_47871705481817
// MI455X (gfx1250) — compile-verified, matrix operand order checked by reading, not run on hardware
//
#include <hip/hip_runtime.h>
#include <hip/hip_bf16.h>
#include <cstdint>
#include <cstddef>

typedef _Float16 half_t;
typedef __attribute__((ext_vector_type(16))) _Float16 v16h;
typedef __attribute__((ext_vector_type(8)))  _Float16 v8h;
typedef __attribute__((ext_vector_type(8)))  float    v8f;

union V16U { v16h v; v8h h[2]; };

#define NB   256
#define HH   15
#define WWD  15
#define NPIX 225
#define PPIX 240
#define CMCH 128
#define CFCH 64
#define XROW 136   // 128 + 8 halves pad -> 272B row stride, conflict-free

// ---- gfx1250 async global->LDS path (guarded; falls back to vector copy) ----
#if defined(__AMDGCN__) && __has_builtin(__builtin_amdgcn_global_load_async_to_lds_b128)
#define ASYNC_LDS 1
typedef int int4v __attribute__((ext_vector_type(4)));
typedef __attribute__((address_space(1))) int4v as1_int4;
typedef __attribute__((address_space(3))) int4v as3_int4;
__device__ inline void async_copy16(const half_t* gsrc, half_t* ldst){
  __builtin_amdgcn_global_load_async_to_lds_b128(
      (as1_int4*)(uintptr_t)gsrc, (as3_int4*)(uintptr_t)ldst, 0, 0);
}
__device__ inline void async_wait0(){
#if __has_builtin(__builtin_amdgcn_s_wait_asynccnt)
  __builtin_amdgcn_s_wait_asynccnt(0);
#else
  asm volatile("s_wait_asynccnt 0" ::: "memory");
#endif
}
#endif

__constant__ int TAPD[4][3][2] = {
  {{0,-1},{0,0},{0,1}},
  {{-1,0},{0,0},{1,0}},
  {{-1,-1},{0,0},{1,1}},
  {{1,-1},{0,0},{-1,1}}};

__device__ inline float silu_f(float x){ return x / (1.0f + __expf(-x)); }

__device__ inline float fq(float x, float s, float qmin, float qmax, bool fl){
  float y = x * s;
  y = fminf(fmaxf(y, qmin), qmax);
  y = fl ? floorf(y) : rintf(y);
  return y / s;
}
__device__ inline float fq8(float x, bool fl=false){ return fq(x, 128.f, -128.f, 127.f, fl); }
__device__ inline float fq16s(float x, float s, bool fl=false){ return fq(x, s, -32768.f, 32767.f, fl); }
__device__ inline float fq32s(float x, float s, bool fl=false){ return fq(x, s, -2.147483648e9f, 2.1474836e9f, fl); }

// ---------------- weight packing: f32 [Co][Ci] row-major -> WMMA B fragments ----
__global__ void pack_mat_kernel(const float* __restrict__ W, half_t* __restrict__ dst,
                                int Co, int Ci){
  int nNT = Co >> 4;
  int bx  = blockIdx.x;              // kt*nNT + nt
  int t = threadIdx.x;               // 0..511
  int lane = t >> 4, j = t & 15;
  int kt = bx / nNT;
  int kb = (lane < 16) ? 0 : 8;
  int K  = kt*32 + ((j < 8) ? (kb + j) : (16 + kb + (j - 8)));
  int n  = (bx % nNT)*16 + (lane & 15);
  dst[((size_t)bx*32 + lane)*16 + j] = (half_t)W[(size_t)n*Ci + K];
}

// ---------------- initial directional conv (Ci=2), scalar ----------------------
__global__ void initial_kernel(const float* __restrict__ board,
                               const float* __restrict__ dw, const float* __restrict__ db,
                               half_t* __restrict__ xout, int dir){
  int p = blockIdx.x % PPIX;
  int b = blockIdx.x / PPIX;
  int co = threadIdx.x;
  size_t oidx = ((size_t)b*PPIX + p)*CMCH + co;
  if (p >= NPIX){ xout[oidx] = (half_t)0.f; return; }
  int h = p / WWD, w = p % WWD;
  float s = db[co];
  #pragma unroll
  for (int k = 0; k < 3; ++k){
    int hh = h + TAPD[dir][k][0], ww = w + TAPD[dir][k][1];
    if (hh < 0 || hh >= HH || ww < 0 || ww >= WWD) continue;
    const float* bp = board + (size_t)b*2*NPIX + hh*WWD + ww;
    s += dw[(k*CMCH + co)*2 + 0] * bp[0];
    s += dw[(k*CMCH + co)*2 + 1] * bp[NPIX];
  }
  xout[oidx] = (half_t)silu_f(s);
}

// ---------------- fused residual block: silu(dconv) -> silu(1x1) + x -----------
// ntaps==3: directional conv block; ntaps==1: Conv0d res block.
__global__ __launch_bounds__(256) void tower_kernel(
    const half_t* __restrict__ xin, half_t* __restrict__ xout,
    const half_t* __restrict__ wd, const float* __restrict__ db,
    const half_t* __restrict__ wc, const float* __restrict__ cb,
    int ntaps, int dir){
  __shared__ __align__(16) half_t lds_xs[3*16*XROW];
  __shared__ __align__(16) half_t lds_y[16*XROW];
  int mt = blockIdx.x % 15;
  int b  = blockIdx.x / 15;
  int tid = threadIdx.x;
  int center = (ntaps == 3) ? 1 : 0;

  // stage shifted rows of x into LDS: 16B segments, async DMA where available
  for (int u = tid; u < ntaps*256; u += 256){
    int tap = u >> 8, rem = u & 255;
    int m = rem >> 4, seg = rem & 15;
    int pix = mt*16 + m;
    const half_t* src = nullptr;
    if (pix < NPIX){
      int h = pix / WWD, w = pix % WWD;
      int di = (ntaps == 1) ? 0 : TAPD[dir][tap][0];
      int dj = (ntaps == 1) ? 0 : TAPD[dir][tap][1];
      int hh = h + di, wp = w + dj;
      if (hh >= 0 && hh < HH && wp >= 0 && wp < WWD)
        src = xin + ((size_t)b*PPIX + hh*WWD + wp)*CMCH + seg*8;
    }
    half_t* dst = &lds_xs[(tap*16 + m)*XROW + seg*8];
    if (src){
#if defined(ASYNC_LDS)
      async_copy16(src, dst);
#else
      *(v8h*)dst = *(const v8h*)src;
#endif
    } else {
      v8h z = {(_Float16)0.f,(_Float16)0.f,(_Float16)0.f,(_Float16)0.f,
               (_Float16)0.f,(_Float16)0.f,(_Float16)0.f,(_Float16)0.f};
      *(v8h*)dst = z;
    }
  }
#if defined(ASYNC_LDS)
  async_wait0();
#endif
  __syncthreads();

  int lane = tid & 31, nt = tid >> 5;      // 8 waves => 8 N-tiles of 16
  int mrow = lane & 15;
  int kb   = (lane < 16) ? 0 : 8;
  int nglob = nt*16 + mrow;                 // C/D layout: lane column
  int mbase = (lane < 16) ? 0 : 8;

  v8f acc = {0.f,0.f,0.f,0.f,0.f,0.f,0.f,0.f};
  for (int tap = 0; tap < ntaps; ++tap){
    #pragma unroll
    for (int kt = 0; kt < 4; ++kt){
      const half_t* ab = &lds_xs[(tap*16 + mrow)*XROW + kt*32];
      V16U a; a.h[0] = *(const v8h*)(ab + kb); a.h[1] = *(const v8h*)(ab + 16 + kb);
      v16h bm = *(const v16h*)(wd + (((size_t)(tap*4 + kt)*8 + nt)*32 + lane)*16);
      acc = __builtin_amdgcn_wmma_f32_16x16x32_f16(false, a.v, false, bm,
                                                   (short)0, acc, false, false);
    }
  }
  float bias = db[nglob];
  #pragma unroll
  for (int r = 0; r < 8; ++r){
    int m2 = r + mbase;
    lds_y[m2*XROW + nglob] = (half_t)silu_f(acc[r] + bias);
  }
  __syncthreads();

  v8f acc2 = {0.f,0.f,0.f,0.f,0.f,0.f,0.f,0.f};
  #pragma unroll
  for (int kt = 0; kt < 4; ++kt){
    const half_t* ab = &lds_y[mrow*XROW + kt*32];
    V16U a; a.h[0] = *(const v8h*)(ab + kb); a.h[1] = *(const v8h*)(ab + 16 + kb);
    v16h bm = *(const v16h*)(wc + (((size_t)kt*8 + nt)*32 + lane)*16);
    acc2 = __builtin_amdgcn_wmma_f32_16x16x32_f16(false, a.v, false, bm,
                                                  (short)0, acc2, false, false);
  }
  float bias2 = cb[nglob];
  #pragma unroll
  for (int r = 0; r < 8; ++r){
    int m2 = r + mbase;
    int pix = mt*16 + m2;
    float v = silu_f(acc2[r] + bias2) + (float)lds_xs[(center*16 + m2)*XROW + nglob];
    xout[((size_t)b*PPIX + pix)*CMCH + nglob] = (pix < NPIX) ? (half_t)v : (half_t)0.f;
  }
}

// ---------------- final 128->64 projection (no act, no residual) ---------------
__global__ __launch_bounds__(128) void final_kernel(
    const half_t* __restrict__ xin, float* __restrict__ fdir,
    const half_t* __restrict__ wf, const float* __restrict__ fb, int dir){
  int mt = blockIdx.x % 15;
  int b  = blockIdx.x / 15;
  int tid = threadIdx.x;
  int lane = tid & 31, nt = tid >> 5;   // 4 waves => N=64
  int mrow = lane & 15;
  int kb   = (lane < 16) ? 0 : 8;
  int nglob = nt*16 + mrow;
  int pix  = mt*16 + mrow;
  bool valid = pix < NPIX;
  const half_t* arow = xin + ((size_t)b*PPIX + pix)*CMCH;
  v8f acc = {0.f,0.f,0.f,0.f,0.f,0.f,0.f,0.f};
  #pragma unroll
  for (int kt = 0; kt < 4; ++kt){
    V16U a;
    if (valid){
      a.h[0] = *(const v8h*)(arow + kt*32 + kb);
      a.h[1] = *(const v8h*)(arow + kt*32 + 16 + kb);
    } else {
      #pragma unroll
      for (int q = 0; q < 16; ++q) a.v[q] = (_Float16)0.f;
    }
    v16h bm = *(const v16h*)(wf + (((size_t)kt*4 + nt)*32 + lane)*16);
    acc = __builtin_amdgcn_wmma_f32_16x16x32_f16(false, a.v, false, bm,
                                                 (short)0, acc, false, false);
  }
  float bias = fb[nglob];
  int mbase = (lane < 16) ? 0 : 8;
  #pragma unroll
  for (int r = 0; r < 8; ++r){
    int p2 = mt*16 + r + mbase;
    fdir[(((size_t)dir*NB + b)*PPIX + p2)*CFCH + nglob] = acc[r] + bias;
  }
}

// ---------------- feature combine: clip, q16/32, relu(mean over dirs) ----------
__global__ void combine_kernel(const float* __restrict__ fdir, float* __restrict__ feat){
  int idx = blockIdx.x*blockDim.x + threadIdx.x;
  if (idx >= NB*CFCH*NPIX) return;
  int p = idx % NPIX; int c = (idx / NPIX) % CFCH; int b = idx / (NPIX*CFCH);
  float s = 0.f;
  #pragma unroll
  for (int d = 0; d < 4; ++d){
    float v = fdir[(((size_t)d*NB + b)*PPIX + p)*CFCH + c];
    v = fminf(fmaxf(v, -16.f), 16.f);
    s += fq16s(v, 32.f);
  }
  feat[idx] = fmaxf(0.25f*s, 0.f);
}

// ---------------- quantized depthwise 3x3 head ---------------------------------
__global__ void dwconv_kernel(const float* __restrict__ feat,
                              const float* __restrict__ w, const float* __restrict__ bias,
                              float* __restrict__ qfeat){
  int idx = blockIdx.x*blockDim.x + threadIdx.x;
  if (idx >= NB*CFCH*NPIX) return;
  int p = idx % NPIX; int c = (idx / NPIX) % CFCH; int b = idx / (NPIX*CFCH);
  if (c < 32){
    int h = p / WWD, wc = p % WWD;
    float s = 0.f;
    #pragma unroll
    for (int ki = 0; ki < 3; ++ki)
      #pragma unroll
      for (int kj = 0; kj < 3; ++kj){
        int hh = h + ki - 1, wp = wc + kj - 1;
        float x = 0.f;
        if (hh >= 0 && hh < HH && wp >= 0 && wp < WWD)
          x = feat[((size_t)b*CFCH + c)*NPIX + hh*WWD + wp];
        float fd = fq16s(x*4.f, 128.f);
        float wq = fq16s(w[c*9 + ki*3 + kj], 65536.f);
        s += fq32s(fd*wq, 8388608.f, true);
      }
    s += fq16s(bias[c], 128.f);
    qfeat[idx] = fq16s(fmaxf(s, 0.f), 128.f);
  } else {
    qfeat[idx] = fq16s(feat[idx], 128.f);
  }
}

// ---------------- spatial / region sums ----------------------------------------
__global__ void sums_kernel(const float* __restrict__ qfeat,
                            float* __restrict__ vbuf, float* __restrict__ regs){
  int b = blockIdx.x; int c = threadIdx.x;
  float tot = 0.f; float rg[3][3] = {{0.f,0.f,0.f},{0.f,0.f,0.f},{0.f,0.f,0.f}};
  const float* base = qfeat + ((size_t)b*CFCH + c)*NPIX;
  for (int p = 0; p < NPIX; ++p){
    float v = base[p];
    tot += v;
    rg[(p / WWD) / 5][(p % WWD) / 5] += v;
  }
  vbuf[(size_t)b*320 + c] = fq32s(tot/256.f, 128.f, true);
  for (int i = 0; i < 3; ++i)
    for (int j = 0; j < 3; ++j)
      regs[((size_t)b*9 + i*3 + j)*CFCH + c] = fq32s(rg[i][j]/32.f, 128.f, true);
}

// ---------------- generic quantized linear -------------------------------------
__global__ void linear_q_kernel(const float* __restrict__ x, int xstride,
                                const float* __restrict__ w, const float* __restrict__ bias,
                                float* __restrict__ y, int ystride,
                                int In, int Out, int relu){
  int b = blockIdx.x; int o = threadIdx.x;
  if (o >= Out) return;
  float s = fq32s(bias[o], 16384.f);
  const float* xr = x + (size_t)b*xstride;
  const float* wr = w + (size_t)o*In;
  for (int i = 0; i < In; ++i)
    s += fq8(xr[i]) * fq8(wr[i]);
  if (relu) s = fmaxf(s, 0.f);
  y[(size_t)b*ystride + o] = s;
}

// ---------------- star block (fused) -------------------------------------------
__global__ __launch_bounds__(128) void star_kernel(
    const float* __restrict__ x, int xstride,
    const float* __restrict__ u1w, const float* __restrict__ u1b,
    const float* __restrict__ u2w, const float* __restrict__ u2b,
    const float* __restrict__ dw, const float* __restrict__ db,
    float* __restrict__ out, int ostride){
  __shared__ float sx1[128], sx2[128], sy[64];
  int b = blockIdx.x; int t = threadIdx.x;
  const float* xr = x + (size_t)b*xstride;
  float s1 = fq32s(u1b[t], 16384.f), s2 = fq32s(u2b[t], 16384.f);
  for (int i = 0; i < 64; ++i){
    float xv = fq8(xr[i]);
    s1 += xv * fq8(u1w[t*64 + i]);
    s2 += xv * fq8(u2w[t*64 + i]);
  }
  sx1[t] = fq8(fmaxf(s1, 0.f), true);
  sx2[t] = fq8(s2, true);
  __syncthreads();
  if (t < 64)
    sy[t] = fq8(sx1[2*t]*sx2[2*t] + sx1[2*t+1]*sx2[2*t+1], true);
  __syncthreads();
  if (t < 64){
    float s = fq32s(db[t], 16384.f);
    for (int i = 0; i < 64; ++i) s += fq8(sy[i]) * fq8(dw[t*64 + i]);
    out[(size_t)b*ostride + t] = fmaxf(s, 0.f);
  }
}

__global__ void avg4_kernel(const float* __restrict__ a, const float* __restrict__ b,
                            const float* __restrict__ c, const float* __restrict__ d,
                            float* __restrict__ out, int n){
  int i = blockIdx.x*blockDim.x + threadIdx.x;
  if (i >= n) return;
  float ab = fq8(0.5f*(a[i] + b[i]), true);
  float cd = fq8(0.5f*(c[i] + d[i]), true);
  out[i] = fq8(0.5f*(ab + cd), true);
}

// ---------------- dynamic per-sample policy conv -------------------------------
__global__ void policy_kernel(const float* __restrict__ qfeat, const float* __restrict__ pw,
                              const float* __restrict__ pout_w, const float* __restrict__ pout_b,
                              float* __restrict__ out){
  int idx = blockIdx.x*blockDim.x + threadIdx.x;
  if (idx >= NB*NPIX) return;
  int p = idx % NPIX, b = idx / NPIX;
  const float* pwr = pw + (size_t)b*528;
  float acc = pout_b[0];
  for (int m = 0; m < 16; ++m){
    float s = 0.f;
    for (int c = 0; c < 32; ++c){
      float wv = fq16s(pwr[m*32 + c], 16384.f, true);
      s += wv * qfeat[((size_t)b*CFCH + c)*NPIX + p];
    }
    s = fmaxf(s + fq32s(pwr[512 + m], 2097152.f, true), 0.f);
    acc += s * pout_w[m];
  }
  out[(size_t)b*NPIX + p] = acc;
}

// -------------------------------------------------------------------------------
extern "C" void kernel_launch(void* const* d_in, const int* in_sizes, int n_in,
                              void* d_out, int out_size, void* d_ws, size_t ws_size,
                              hipStream_t stream){
  (void)in_sizes; (void)n_in; (void)out_size; (void)ws_size;
  const float* board = (const float*)d_in[0];
  auto F = [&](int i){ return (const float*)d_in[i]; };

  uint8_t* ws = (uint8_t*)d_ws;
  size_t off = 0;
  auto alloc = [&](size_t bytes)->uint8_t*{
    uint8_t* p = ws + off;
    off += (bytes + 255) & ~(size_t)255;
    return p;
  };
  half_t* xA     = (half_t*)alloc(sizeof(half_t)*(size_t)NB*PPIX*CMCH);
  half_t* xB     = (half_t*)alloc(sizeof(half_t)*(size_t)NB*PPIX*CMCH);
  half_t* packed = (half_t*)alloc(sizeof(half_t)*2*303104);
  float*  fdir   = (float*)alloc(sizeof(float)*(size_t)4*NB*PPIX*CFCH);
  float*  feat   = (float*)alloc(sizeof(float)*(size_t)NB*CFCH*NPIX);
  float*  qfeat  = (float*)alloc(sizeof(float)*(size_t)NB*CFCH*NPIX);
  float*  vbuf   = (float*)alloc(sizeof(float)*NB*320);
  float*  regs   = (float*)alloc(sizeof(float)*NB*9*CFCH);
  float*  pw1o   = (float*)alloc(sizeof(float)*NB*64);
  float*  pw2o   = (float*)alloc(sizeof(float)*NB*528);
  float*  sv     = (float*)alloc(sizeof(float)*9*NB*64);
  float*  qv     = (float*)alloc(sizeof(float)*4*NB*64);
  float*  t1     = (float*)alloc(sizeof(float)*NB*64);
  float*  t2     = (float*)alloc(sizeof(float)*NB*64);

  // -------- pack mapping weights into WMMA B-fragment layout --------
  // flat input indexing follows JAX pytree (sorted dict keys):
  // map1 base=3, map2 base=27; per block: +0 cb, +1 cw, +2 dcb, +3 dcw;
  // +16 c0_1b, +17 c0_1w, +18 c0_2b, +19 c0_2w, +20 db, +21 dw, +22 fb, +23 fw
  for (int m = 0; m < 2; ++m){
    int base = (m == 0) ? 3 : 27;
    half_t* P = packed + (size_t)m*303104;
    for (int blk = 0; blk < 4; ++blk){
      const float* dcw = F(base + 4*blk + 3);
      for (int tap = 0; tap < 3; ++tap)
        pack_mat_kernel<<<32, 512, 0, stream>>>(dcw + (size_t)tap*CMCH*CMCH,
                                                P + blk*65536 + tap*16384, CMCH, CMCH);
      pack_mat_kernel<<<32, 512, 0, stream>>>(F(base + 4*blk + 1),
                                              P + blk*65536 + 49152, CMCH, CMCH);
    }
    pack_mat_kernel<<<32, 512, 0, stream>>>(F(base + 17), P + 262144, CMCH, CMCH);
    pack_mat_kernel<<<32, 512, 0, stream>>>(F(base + 19), P + 278528, CMCH, CMCH);
    pack_mat_kernel<<<16, 512, 0, stream>>>(F(base + 23), P + 294912, CFCH, CMCH);
  }

  // -------- mapping towers, one direction at a time (WMMA) --------
  for (int dir = 0; dir < 4; ++dir){
    int m = dir >> 1;
    int base = (m == 0) ? 3 : 27;
    half_t* P = packed + (size_t)m*303104;
    initial_kernel<<<NB*PPIX, CMCH, 0, stream>>>(board, F(base+21), F(base+20), xA, dir);
    half_t* cur = xA; half_t* nxt = xB;
    for (int blk = 0; blk < 4; ++blk){
      tower_kernel<<<NB*15, 256, 0, stream>>>(cur, nxt,
          P + blk*65536,        F(base + 4*blk + 2),
          P + blk*65536 + 49152, F(base + 4*blk + 0), 3, dir);
      half_t* tmp = cur; cur = nxt; nxt = tmp;
    }
    tower_kernel<<<NB*15, 256, 0, stream>>>(cur, nxt,
        P + 262144, F(base + 16), P + 278528, F(base + 18), 1, dir);
    { half_t* tmp = cur; cur = nxt; nxt = tmp; }
    final_kernel<<<NB*15, 128, 0, stream>>>(cur, fdir, P + 294912, F(base + 22), dir);
  }

  // -------- heads --------
  int tot = NB*CFCH*NPIX;
  combine_kernel<<<(tot + 255)/256, 256, 0, stream>>>(fdir, feat);
  dwconv_kernel<<<(tot + 255)/256, 256, 0, stream>>>(feat, F(2), F(1), qfeat);
  sums_kernel<<<NB, 64, 0, stream>>>(qfeat, vbuf, regs);

  // policy: pw1 (relu) -> pw2 -> dynamic conv
  linear_q_kernel<<<NB, 64, 0, stream>>>(vbuf, 320, F(54), F(53), pw1o, 64, 64, 64, 1);
  linear_q_kernel<<<NB, 528, 0, stream>>>(pw1o, 64, F(56), F(55), pw2o, 528, 64, 528, 0);
  policy_kernel<<<(NB*NPIX + 255)/256, 256, 0, stream>>>(
      qfeat, pw2o, F(52), F(51), (float*)d_out + NB*3);

  // value: 9 region star blocks -> 4 averaged quadrants -> vq stars -> MLP
  auto starcall = [&](int sb, const float* x, int xstride, float* out, int ostride){
    star_kernel<<<NB, 128, 0, stream>>>(x, xstride,
        F(sb+3), F(sb+2), F(sb+5), F(sb+4), F(sb+1), F(sb+0), out, ostride);
  };
  const int VC = 57, VE = 63, VM = 75, VQ = 81;
  int n64 = NB*64;
  int corner[4] = {0, 2, 6, 8};
  int edge[4]   = {1, 3, 5, 7};
  for (int q = 0; q < 4; ++q)
    starcall(VC, regs + corner[q]*64, 9*64, sv + (size_t)corner[q]*n64, 64);
  for (int q = 0; q < 4; ++q)
    starcall(VE, regs + edge[q]*64, 9*64, sv + (size_t)edge[q]*n64, 64);
  starcall(VM, regs + 4*64, 9*64, sv + (size_t)4*n64, 64);

  int qr[4][4] = {{0,1,3,4},{1,2,4,5},{3,4,6,7},{4,5,7,8}};
  for (int q = 0; q < 4; ++q)
    avg4_kernel<<<(n64 + 255)/256, 256, 0, stream>>>(
        sv + (size_t)qr[q][0]*n64, sv + (size_t)qr[q][1]*n64,
        sv + (size_t)qr[q][2]*n64, sv + (size_t)qr[q][3]*n64,
        qv + (size_t)q*n64, n64);
  for (int q = 0; q < 4; ++q)
    starcall(VQ, qv + (size_t)q*n64, 64, vbuf + 64 + q*64, 320);

  linear_q_kernel<<<NB, 64, 0, stream>>>(vbuf, 320, F(70), F(69), t1, 64, 320, 64, 1);
  linear_q_kernel<<<NB, 64, 0, stream>>>(t1, 64, F(72), F(71), t2, 64, 64, 64, 1);
  linear_q_kernel<<<NB, 3, 0, stream>>>(t2, 64, F(74), F(73), (float*)d_out, 3, 64, 3, 0);
}
